// Model_43516608643518
// MI455X (gfx1250) — matrix operand order repaired and checked by reading, not compiled, not run on hardware
//
#include <hip/hip_runtime.h>

// ---------------- problem constants ----------------
#define BB   64      // batch
#define LL   512     // sequence length
#define FF   256     // input features
#define HD   1024    // hidden dim
#define TT   128     // output classes
#define NWG1 64      // layer-1 workgroups (16 cols each)
#define NWG2 64      // layer-2 workgroups (16 cols each)
#define NWGT (NWG1 + NWG2)
#define LDSTRIDE (HD + 8)   // LDS row stride in elements: 16B pad -> optimal 2-cycle banks

// ---------------- WMMA types ----------------
typedef __attribute__((ext_vector_type(16))) __bf16 v16bf;
typedef __attribute__((ext_vector_type(8)))  float  v8f;

union Frag { v16bf v; uint4 q[2]; };

// ---------------- scalar helpers ----------------
__device__ __forceinline__ unsigned short f32_to_bf16(float f) {
  unsigned u = __float_as_uint(f);
  u += 0x7FFFu + ((u >> 16) & 1u);          // round-to-nearest-even
  return (unsigned short)(u >> 16);
}
__device__ __forceinline__ float bf16_to_f32(unsigned short h) {
  return __uint_as_float(((unsigned)h) << 16);
}

// A operand: 16x32 bf16 tile from row-major [rows][K] (K contiguous).
__device__ __forceinline__ v16bf load_a(const unsigned short* __restrict__ base,
                                        int arow, int k0, int ld, int half) {
  Frag f;
  const unsigned short* p = base + (size_t)arow * ld + k0 + half * 16;
  f.q[0] = *(const uint4*)(p);
  f.q[1] = *(const uint4*)(p + 8);
  return f.v;
}
// B operand (global): 32x16 (KxN) from weight stored [N][K] row-major.
__device__ __forceinline__ v16bf load_b(const unsigned short* __restrict__ w,
                                        int bcol, int k0, int ld, int half) {
  Frag f;
  const unsigned short* p = w + (size_t)bcol * ld + k0 + half * 16;
  f.q[0] = *(const uint4*)(p);
  f.q[1] = *(const uint4*)(p + 8);
  return f.v;
}
// B operand from LDS slice [16][LDSTRIDE] (per-WG 16-column weight slice)
__device__ __forceinline__ v16bf load_b_lds(const unsigned short* s,
                                            int rr, int k0, int half) {
  Frag f;
  const unsigned short* p = s + rr * LDSTRIDE + k0 + half * 16;
  f.q[0] = *(const uint4*)(p);
  f.q[1] = *(const uint4*)(p + 8);
  return f.v;
}
__device__ __forceinline__ v8f wmma_bf16(v16bf a, v16bf b, v8f c) {
  return __builtin_amdgcn_wmma_f32_16x16x32_bf16(false, a, false, b, (short)0, c,
                                                 false, false);
}

// ---------------- grid barrier ----------------
__device__ __forceinline__ void grid_barrier(unsigned* cnt, unsigned* gen) {
  __threadfence();
  __syncthreads();
  if (threadIdx.x == 0) {
    unsigned g = __hip_atomic_load(gen, __ATOMIC_RELAXED, __HIP_MEMORY_SCOPE_AGENT);
    unsigned old = __hip_atomic_fetch_add(cnt, 1u, __ATOMIC_ACQ_REL, __HIP_MEMORY_SCOPE_AGENT);
    if (old == NWGT - 1) {
      __hip_atomic_store(cnt, 0u, __ATOMIC_RELAXED, __HIP_MEMORY_SCOPE_AGENT);
      __hip_atomic_fetch_add(gen, 1u, __ATOMIC_RELEASE, __HIP_MEMORY_SCOPE_AGENT);
    } else {
      while (__hip_atomic_load(gen, __ATOMIC_ACQUIRE, __HIP_MEMORY_SCOPE_AGENT) == g)
        __builtin_amdgcn_s_sleep(1);
    }
  }
  __syncthreads();
}

// ---------------- prep kernels ----------------
// x[b][t][f] (f32) -> x_bf[t][b][f] (bf16)
__global__ void convert_x_kernel(const float* __restrict__ x,
                                 unsigned short* __restrict__ x_bf) {
  int idx = blockIdx.x * blockDim.x + threadIdx.x;
  if (idx >= LL * BB * FF) return;
  int f = idx % FF;
  int tb = idx / FF;
  int b = tb % BB;
  int t = tb / BB;
  x_bf[idx] = f32_to_bf16(x[((size_t)b * LL + t) * FF + f]);
}

__global__ void convert_w_kernel(const float* __restrict__ W_ih,
                                 const float* __restrict__ b_ih,
                                 const float* __restrict__ W_hh,
                                 const float* __restrict__ b_hh,
                                 unsigned short* __restrict__ Whh1,
                                 unsigned short* __restrict__ Wih2,
                                 unsigned short* __restrict__ Whh2,
                                 float* __restrict__ b2,
                                 unsigned short* __restrict__ h1buf,
                                 unsigned short* __restrict__ h2buf,
                                 unsigned* __restrict__ bar) {
  int idx = blockIdx.x * blockDim.x + threadIdx.x;
  if (idx >= HD * HD) return;
  Whh1[idx] = f32_to_bf16(W_hh[idx]);
  Wih2[idx] = f32_to_bf16(W_ih[(size_t)HD * HD + idx]);
  Whh2[idx] = f32_to_bf16(W_hh[(size_t)HD * HD + idx]);
  if (idx < HD) b2[idx] = b_ih[HD + idx] + b_hh[HD + idx];
  if (idx < 2 * BB * HD) { h1buf[idx] = 0; h2buf[idx] = 0; }
  if (idx < 8) bar[idx] = 0;
}

// Wc1 = W_ih[0] @ W_in (fold input projection into layer 1) + fused bias
__global__ void make_wc1_kernel(const float* __restrict__ W_in,
                                const float* __restrict__ b_in,
                                const float* __restrict__ W_ih,
                                const float* __restrict__ b_ih,
                                const float* __restrict__ b_hh,
                                unsigned short* __restrict__ Wc1,
                                float* __restrict__ bc1) {
  int n = blockIdx.x;      // 0..HD-1
  int f = threadIdx.x;     // 0..FF-1
  const float* wrow = W_ih + (size_t)n * HD;
  float acc = 0.f;
  for (int k = 0; k < HD; ++k)
    acc = fmaf(wrow[k], W_in[(size_t)k * FF + f], acc);
  Wc1[(size_t)n * FF + f] = f32_to_bf16(acc);
  if (f == 0) {
    float bb = b_ih[n] + b_hh[n];
    for (int k = 0; k < HD; ++k) bb += wrow[k] * b_in[k];
    bc1[n] = bb;
  }
}

// ---------------- persistent layer-pipelined RNN ----------------
// 128 WGs x 128 threads. WGs [0,64): layer 1 (h1[k] at tick k).
// WGs [64,128): layer 2 (h2[k-1] at tick k). One grid barrier per tick.
// h1[k], h2[k] stored at parity k&1 of their ping-pong buffers.
__global__ void __launch_bounds__(128)
rnn_kernel(const unsigned short* __restrict__ x_bf,   // [LL][BB][FF]
           const unsigned short* __restrict__ Wc1,    // [HD][FF]
           const unsigned short* __restrict__ Whh1,   // [HD][HD]
           const unsigned short* __restrict__ Wih2,   // [HD][HD]
           const unsigned short* __restrict__ Whh2,   // [HD][HD]
           const float* __restrict__ bc1,
           const float* __restrict__ b2,
           unsigned short* __restrict__ h1buf,        // [2][BB][HD]
           unsigned short* __restrict__ h2buf,        // [2][BB][HD]
           unsigned* bar_cnt, unsigned* bar_gen,
           const int* __restrict__ pred_len_p) {
  __shared__ unsigned short smem[16 * LDSTRIDE];      // 33 KB weight slice

  const int tend = LL - pred_len_p[0];                // #recurrence steps (488)
  const bool is_l2 = (blockIdx.x >= NWG1);
  const int g    = is_l2 ? (blockIdx.x - NWG1) : blockIdx.x;  // 0..63
  const int tid  = threadIdx.x;
  const int w    = tid >> 5;                          // wave 0..3 -> M tile
  const int lane = tid & 31;
  const int r    = lane & 15;
  const int half = lane >> 4;
  const int m0   = w * 16;
  const int n0   = g * 16;                            // global column base
  const int arow = m0 + r;
  const int bcol = n0 + r;

  // stage this WG's LDS weight slice: layer1 -> Whh1 rows, layer2 -> Wih2 rows
  {
    const unsigned short* src = is_l2 ? Wih2 : Whh1;
    for (int i = tid; i < 16 * (HD / 8); i += 128) {  // 16 rows x 128 uint4
      int row = i >> 7;
      int c   = i & 127;
      *(uint4*)(&smem[row * LDSTRIDE + c * 8]) =
          *(const uint4*)(src + (size_t)(n0 + row) * HD + c * 8);
    }
  }
  __syncthreads();

  const float bias = is_l2 ? b2[bcol] : bc1[bcol];

  for (int k = 0; k <= tend; ++k) {
    if (!is_l2) {
      // ---- layer 1, tick k: h1[k] = tanh(x_k @ Wc1^T + h1[k-1] @ Whh1^T + b)
      if (k < tend) {
        const unsigned short* h1p = h1buf + (size_t)((k + 1) & 1) * (BB * HD); // h1[k-1]
        unsigned short*       h1c = h1buf + (size_t)(k & 1) * (BB * HD);       // h1[k]
        const unsigned short* xt  = x_bf + (size_t)k * (BB * FF);
        v8f acc = {bias, bias, bias, bias, bias, bias, bias, bias};
#pragma unroll
        for (int k0 = 0; k0 < FF; k0 += 32)   // Wc1 frags get hoisted to VGPRs
          acc = wmma_bf16(load_a(xt, arow, k0, FF, half),
                          load_b(Wc1, bcol, k0, FF, half), acc);
#pragma unroll 8
        for (int k0 = 0; k0 < HD; k0 += 32)   // Whh1 from LDS
          acc = wmma_bf16(load_a(h1p, arow, k0, HD, half),
                          load_b_lds(smem, r, k0, half), acc);
        if (k + 1 < tend)
          __builtin_prefetch(x_bf + (size_t)(k + 1) * (BB * FF) + (size_t)arow * FF, 0, 1);
#pragma unroll
        for (int j = 0; j < 8; ++j) {
          int row = m0 + j + half * 8;        // D layout: VGPR j -> M=j / M=8+j
          h1c[(size_t)row * HD + bcol] = f32_to_bf16(tanhf(acc[j]));
        }
      }
    } else {
      // ---- layer 2, tick k: h2[k-1] = tanh(h1[k-1] @ Wih2^T + h2[k-2] @ Whh2^T + b)
      if (k > 0) {
        const unsigned short* h1v = h1buf + (size_t)((k + 1) & 1) * (BB * HD); // h1[k-1]
        const unsigned short* h2p = h2buf + (size_t)(k & 1) * (BB * HD);       // h2[k-2]
        unsigned short*       h2c = h2buf + (size_t)((k + 1) & 1) * (BB * HD); // h2[k-1]
        v8f acc = {bias, bias, bias, bias, bias, bias, bias, bias};
#pragma unroll 8
        for (int k0 = 0; k0 < HD; k0 += 32) { // Wih2 from LDS, Whh2 from L2
          acc = wmma_bf16(load_a(h1v, arow, k0, HD, half),
                          load_b_lds(smem, r, k0, half), acc);
          acc = wmma_bf16(load_a(h2p, arow, k0, HD, half),
                          load_b(Whh2, bcol, k0, HD, half), acc);
        }
#pragma unroll
        for (int j = 0; j < 8; ++j) {
          int row = m0 + j + half * 8;
          h2c[(size_t)row * HD + bcol] = f32_to_bf16(tanhf(acc[j]));
        }
      }
    }
    grid_barrier(bar_cnt, bar_gen);           // one barrier per tick
  }
}

// ---------------- head: out[b][t] = relu(h2[tend-1][b]) . W_out[t] + b_out[t]
__global__ void head_kernel(const unsigned short* __restrict__ h2buf,
                            const float* __restrict__ W_out,
                            const float* __restrict__ b_out,
                            float* __restrict__ out,
                            const int* __restrict__ pred_len_p) {
  const int tend = LL - pred_len_p[0];
  const unsigned short* h = h2buf + (size_t)((tend - 1) & 1) * (BB * HD);
  int b = blockIdx.x;       // 0..BB-1
  int t = threadIdx.x;      // 0..TT-1
  float acc = b_out[t];
  for (int k = 0; k < HD; ++k) {
    float hv = bf16_to_f32(h[(size_t)b * HD + k]);
    hv = hv > 0.f ? hv : 0.f;
    acc = fmaf(hv, W_out[(size_t)t * HD + k], acc);
  }
  out[(size_t)b * TT + t] = acc;
}

// ---------------- launch ----------------
extern "C" void kernel_launch(void* const* d_in, const int* in_sizes, int n_in,
                              void* d_out, int out_size, void* d_ws, size_t ws_size,
                              hipStream_t stream) {
  const float* x      = (const float*)d_in[0];
  const float* W_in   = (const float*)d_in[1];
  const float* b_in   = (const float*)d_in[2];
  const float* W_ih   = (const float*)d_in[3];
  const float* b_ih   = (const float*)d_in[4];
  const float* W_hh   = (const float*)d_in[5];
  const float* b_hh   = (const float*)d_in[6];
  const float* W_out  = (const float*)d_in[7];
  const float* b_out  = (const float*)d_in[8];
  const int*   predlp = (const int*)d_in[9];
  float* out = (float*)d_out;
  (void)in_sizes; (void)n_in; (void)out_size; (void)ws_size;

  char* ws = (char*)d_ws;
  size_t off = 0;
  auto alloc = [&](size_t bytes) -> void* {
    void* p = ws + off;
    off = (off + bytes + 255) & ~(size_t)255;
    return p;
  };
  unsigned short* x_bf  = (unsigned short*)alloc((size_t)LL * BB * FF * 2); // 16 MB
  unsigned short* Wc1   = (unsigned short*)alloc((size_t)HD * FF * 2);
  unsigned short* Whh1  = (unsigned short*)alloc((size_t)HD * HD * 2);
  unsigned short* Wih2  = (unsigned short*)alloc((size_t)HD * HD * 2);
  unsigned short* Whh2  = (unsigned short*)alloc((size_t)HD * HD * 2);
  unsigned short* h1buf = (unsigned short*)alloc((size_t)2 * BB * HD * 2);
  unsigned short* h2buf = (unsigned short*)alloc((size_t)2 * BB * HD * 2);
  float*          bc1   = (float*)alloc((size_t)HD * 4);
  float*          b2    = (float*)alloc((size_t)HD * 4);
  unsigned*       bar   = (unsigned*)alloc(256);

  convert_x_kernel<<<(LL * BB * FF + 255) / 256, 256, 0, stream>>>(x, x_bf);
  convert_w_kernel<<<(HD * HD + 255) / 256, 256, 0, stream>>>(
      W_ih, b_ih, W_hh, b_hh, Whh1, Wih2, Whh2, b2, h1buf, h2buf, bar);
  make_wc1_kernel<<<HD, FF, 0, stream>>>(W_in, b_in, W_ih, b_ih, b_hh, Wc1, bc1);
  rnn_kernel<<<NWGT, 128, 0, stream>>>(x_bf, Wc1, Whh1, Wih2, Whh2, bc1, b2,
                                       h1buf, h2buf, bar, bar + 1, predlp);
  head_kernel<<<BB, TT, 0, stream>>>(h2buf, W_out, b_out, out, predlp);
}
